// SparseMoE_39195871543621
// MI455X (gfx1250) — compile-verified, matrix operand order checked by reading, not run on hardware
//
#include <hip/hip_runtime.h>
#include <hip/hip_bf16.h>

// ---------------- problem constants (match reference) ----------------
#define T_TOK   4096          // B*S tokens
#define D_EMB   768
#define F_HID   3072
#define N_EXP   8
#define MT      32            // tokens per FFN tile
#define XS_STR  (D_EMB + 8)   // LDS stride (halves); 776*2 bytes, 16B aligned rows
#define HS_STR  (F_HID + 8)   // LDS stride (halves); 3080*2 bytes, 16B aligned rows
#define KT1     (D_EMB / 32)  // 24  k-blocks, GEMM1
#define NT1     (F_HID / 16)  // 192 n-tiles,  GEMM1
#define KT2     (F_HID / 32)  // 96  k-blocks, GEMM2
#define NT2     (D_EMB / 16)  // 48  n-tiles,  GEMM2
#define NB      4             // n-tiles per wave per k-sweep (register blocking)
#define GATE_THRESH 1e-9f

typedef __attribute__((ext_vector_type(16))) _Float16 v16h;
typedef __attribute__((ext_vector_type(8)))  _Float16 v8h;
typedef __attribute__((ext_vector_type(8)))  float    v8f;

// ---- A-fragment from LDS --------------------------------------------------
// 16-bit A 16x32 layout (ISA 7.12.2): lane L<16 holds row L, K={0..7,16..23};
// lane L>=16 holds row L-16, K={8..15,24..31}. p is pre-offset by (row, kb+ksel).
__device__ __forceinline__ v16h load_a_lds(const _Float16* p) {
    union { v16h v; v8h h[2]; } u;
    u.h[0] = *(const v8h*)(p);        // 16B ds_load
    u.h[1] = *(const v8h*)(p + 16);   // 16B ds_load
    return u.v;
}

// ---------------- kernel 0: zero output + routing counters -----------------
__global__ void moe_zero(float* __restrict__ out, int n, int* __restrict__ cnt) {
    int i = blockIdx.x * blockDim.x + threadIdx.x;
    if (i < n) out[i] = 0.0f;
    if (i < N_EXP) cnt[i] = 0;
}

// ---------------- kernel P: pack f32 weights -> f16 B-fragment layout ------
// Output layout: [e][nt][kt][lane][16 halves]  (one v16h per (frag,lane)).
// Lane l of fragment (kt,nt): column n = nt*16 + (l&15), K rows
// kt*32 + {ksel..ksel+7, ksel+16..ksel+23} with ksel = (l>=16 ? 8 : 0).
__global__ __launch_bounds__(256) void moe_pack_b(
    const float* __restrict__ W, _Float16* __restrict__ Wh, int K, int N)
{
    const int KT = K >> 5, NT = N >> 4;
    const size_t total = (size_t)N_EXP * NT * KT * 32;
    size_t tid = (size_t)blockIdx.x * blockDim.x + threadIdx.x;
    if (tid >= total) return;

    const int lane = (int)(tid & 31);
    size_t f  = tid >> 5;                 // ((e*NT + nt)*KT + kt)
    const int kt = (int)(f % KT);
    f /= KT;
    const int nt = (int)(f % NT);
    const int e  = (int)(f / NT);

    const float* We = W + (size_t)e * K * N;
    const int n  = (nt << 4) + (lane & 15);
    const int k0 = (kt << 5) + ((lane >> 4) << 3);

    v16h b;
#pragma unroll
    for (int i = 0; i < 8; ++i) b[i]     = (_Float16)We[(size_t)(k0 + i)      * N + n];
#pragma unroll
    for (int i = 0; i < 8; ++i) b[i + 8] = (_Float16)We[(size_t)(k0 + 16 + i) * N + n];
    *((v16h*)Wh + tid) = b;               // 32B contiguous, coalesced per wave
}

// ---------------- kernel 1: noisy top-2 router + compaction ----------------
__global__ __launch_bounds__(256) void moe_router(
    const float* __restrict__ x, const float* __restrict__ noise,
    const float* __restrict__ Wg, const float* __restrict__ bg,
    const float* __restrict__ Wn, const float* __restrict__ bn,
    int* __restrict__ cnt, int* __restrict__ tok_list, float* __restrict__ gate_list)
{
    const int gtid = blockIdx.x * blockDim.x + threadIdx.x;
    const int t    = gtid >> 5;
    const int lane = gtid & 31;
    if (t >= T_TOK) return;

    const float* xr = x + (size_t)t * D_EMB;
    float ag[N_EXP], an[N_EXP];
#pragma unroll
    for (int e = 0; e < N_EXP; ++e) { ag[e] = 0.f; an[e] = 0.f; }

    for (int k = lane; k < D_EMB; k += 32) {
        const float xv = xr[k];
        const float* wg = Wg + (size_t)k * N_EXP;   // Wg is [d][E]
        const float* wn = Wn + (size_t)k * N_EXP;
#pragma unroll
        for (int e = 0; e < N_EXP; ++e) {
            ag[e] = fmaf(xv, wg[e], ag[e]);
            an[e] = fmaf(xv, wn[e], an[e]);
        }
    }
#pragma unroll
    for (int off = 16; off > 0; off >>= 1) {
#pragma unroll
        for (int e = 0; e < N_EXP; ++e) {
            ag[e] += __shfl_down(ag[e], off, 32);
            an[e] += __shfl_down(an[e], off, 32);
        }
    }
    if (lane != 0) return;

    float nv[N_EXP];
#pragma unroll
    for (int e = 0; e < N_EXP; ++e) {
        float nl = an[e] + bn[e];
        float sp = (nl > 20.f) ? nl : log1pf(expf(nl));   // softplus
        nv[e] = ag[e] + bg[e] + noise[(size_t)t * N_EXP + e] * sp;
    }
    int i0 = 0;
#pragma unroll
    for (int e = 1; e < N_EXP; ++e) if (nv[e] > nv[i0]) i0 = e;   // ties -> lowest idx
    int i1 = (i0 == 0) ? 1 : 0;
#pragma unroll
    for (int e = 0; e < N_EXP; ++e) if (e != i0 && nv[e] > nv[i1]) i1 = e;

    const float ex = expf(nv[i1] - nv[i0]);     // <= 1, no overflow
    const float p0 = 1.f / (1.f + ex);
    const float p1 = ex / (1.f + ex);

    int s0 = atomicAdd(&cnt[i0], 1);
    tok_list [i0 * T_TOK + s0] = t;
    gate_list[i0 * T_TOK + s0] = p0;
    if (p1 > GATE_THRESH) {
        int s1 = atomicAdd(&cnt[i1], 1);
        tok_list [i1 * T_TOK + s1] = t;
        gate_list[i1 * T_TOK + s1] = p1;
    }
}

// ---------------- kernel 2: fused gathered expert FFN ----------------------
// Block = (expert e, 32-token tile). 8 waves = 2 M-halves x 4 N-groups.
// Register-blocked 4 n-tiles per k-sweep: per iteration 2x ds_load_b128 (A,
// shared) + 8x global_load_b128 (4 packed-B streams) + 4x WMMA.
__global__ __launch_bounds__(256) void moe_ffn(
    const float* __restrict__ x,
    const _Float16* __restrict__ W1h, const float* __restrict__ b1,
    const _Float16* __restrict__ W2h, const float* __restrict__ b2,
    const int* __restrict__ cnt, const int* __restrict__ tok_list,
    const float* __restrict__ gate_list, float* __restrict__ out)
{
    const int e     = blockIdx.y;
    const int tile0 = blockIdx.x * MT;
    const int n_e   = cnt[e];
    if (tile0 >= n_e) return;                    // block-uniform exit

    extern __shared__ char smem_raw[];
    _Float16* Xs = (_Float16*)smem_raw;                       // [MT][XS_STR]
    _Float16* Hs = ((_Float16*)smem_raw) + MT * XS_STR;       // [MT][HS_STR]
    __shared__ int   toks[MT];
    __shared__ float gts[MT];

    if (threadIdx.x < MT) {
        const int i = tile0 + (int)threadIdx.x;
        if (i < n_e) {
            toks[threadIdx.x] = tok_list [e * T_TOK + i];
            gts [threadIdx.x] = gate_list[e * T_TOK + i];
        } else {
            toks[threadIdx.x] = 0;              // dummy row, gate 0 discards it
            gts [threadIdx.x] = 0.0f;
        }
    }
    __syncthreads();

    // Gather + convert token activations into LDS (float4 loads, 8B LDS stores).
    for (int idx = threadIdx.x; idx < MT * (D_EMB / 4); idx += 256) {
        const int r  = idx / (D_EMB / 4);
        const int c4 = idx - r * (D_EMB / 4);
        const float4 v = ((const float4*)(x + (size_t)toks[r] * D_EMB))[c4];
        _Float16* dst = &Xs[r * XS_STR + (c4 << 2)];
        dst[0] = (_Float16)v.x; dst[1] = (_Float16)v.y;
        dst[2] = (_Float16)v.z; dst[3] = (_Float16)v.w;
    }
    __syncthreads();

    const int lane = threadIdx.x & 31;
    const int wid  = threadIdx.x >> 5;
    const int mh   = wid & 1;                    // M-half: rows 0..15 / 16..31
    const int ng   = wid >> 1;                   // N-group 0..3
    const int row  = (mh << 4) + (lane & 15);    // A-frag row for this lane
    const int ksel = (lane >> 4) << 3;           // 0 / 8 halves (lane-hi K offset)
    const int ncol = lane & 15;                  // B/C column for this lane
    const int mofs = (mh << 4) + ((lane >> 4) << 3);  // C row base for VGPR i

    // ---- GEMM1: H = relu(X @ W1[e] + b1[e]) ----
    const v16h*  W1f = (const v16h*)W1h;
    const float* b1e = b1 + (size_t)e * F_HID;
    for (int nt = ng * NB; nt < NT1; nt += 4 * NB) {
        const v16h* bp = W1f + ((size_t)(e * NT1 + nt) * KT1) * 32 + lane;
        v8f cc[NB] = {};
#pragma unroll 2
        for (int kt = 0; kt < KT1; ++kt) {
            const v16h a = load_a_lds(&Xs[row * XS_STR + (kt << 5) + ksel]);
            const size_t o = (size_t)kt * 32;
            __builtin_prefetch(bp + o + 128, 0, 0);
#pragma unroll
            for (int j = 0; j < NB; ++j) {
                const v16h b = bp[o + (size_t)j * KT1 * 32];   // 32B coalesced
                cc[j] = __builtin_amdgcn_wmma_f32_16x16x32_f16(
                            false, a, false, b, (short)0, cc[j], false, false);
            }
        }
#pragma unroll
        for (int j = 0; j < NB; ++j) {
            const int nb = (nt + j) << 4;
            const float bias = b1e[nb + ncol];
#pragma unroll
            for (int i = 0; i < 8; ++i) {        // C layout: VGPR i -> M = mofs+i
                const float v = fmaxf(cc[j][i] + bias, 0.0f);
                Hs[(mofs + i) * HS_STR + nb + ncol] = (_Float16)v;
            }
        }
    }
    __syncthreads();

    // ---- GEMM2: Y = H @ W2[e] + b2[e] ----
    const v16h*  W2f = (const v16h*)W2h;
    const float* b2e = b2 + (size_t)e * D_EMB;
    for (int nt = ng * NB; nt < NT2; nt += 4 * NB) {
        const v16h* bp = W2f + ((size_t)(e * NT2 + nt) * KT2) * 32 + lane;
        v8f cc[NB] = {};
#pragma unroll 2
        for (int kt = 0; kt < KT2; ++kt) {
            const v16h a = load_a_lds(&Hs[row * HS_STR + (kt << 5) + ksel]);
            const size_t o = (size_t)kt * 32;
            __builtin_prefetch(bp + o + 128, 0, 0);
#pragma unroll
            for (int j = 0; j < NB; ++j) {
                const v16h b = bp[o + (size_t)j * KT2 * 32];
                cc[j] = __builtin_amdgcn_wmma_f32_16x16x32_f16(
                            false, a, false, b, (short)0, cc[j], false, false);
            }
        }
#pragma unroll
        for (int j = 0; j < NB; ++j) {
            const int nb = (nt + j) << 4;
            const float bias = b2e[nb + ncol];
#pragma unroll
            for (int i = 0; i < 8; ++i) {
                const int   m = mofs + i;
                const float g = gts[m];
                if (g != 0.0f) {                 // <=2 commutative adds/element
                    atomicAdd(&out[(size_t)toks[m] * D_EMB + nb + ncol],
                              g * (cc[j][i] + bias));
                }
            }
        }
    }
}

// ---------------- host launcher --------------------------------------------
extern "C" void kernel_launch(void* const* d_in, const int* in_sizes, int n_in,
                              void* d_out, int out_size, void* d_ws, size_t ws_size,
                              hipStream_t stream)
{
    (void)in_sizes; (void)n_in; (void)out_size; (void)ws_size;
    const float* x     = (const float*)d_in[0];
    const float* noise = (const float*)d_in[1];
    const float* Wg    = (const float*)d_in[2];
    const float* bg    = (const float*)d_in[3];
    const float* Wn    = (const float*)d_in[4];
    const float* bn    = (const float*)d_in[5];
    const float* W1    = (const float*)d_in[6];
    const float* b1    = (const float*)d_in[7];
    const float* W2    = (const float*)d_in[8];
    const float* b2    = (const float*)d_in[9];
    float* out = (float*)d_out;

    // ---- workspace layout (~76 MB) ----
    char* ws = (char*)d_ws;
    int*      cnt   = (int*)ws;                                   // 32 B
    int*      toks  = (int*)(ws + 256);                           // 128 KB
    float*    gates = (float*)(ws + 256 + 131072);                // 128 KB
    _Float16* W1h   = (_Float16*)(ws + 262400);                   // 36 MB (f16, fragment-major)
    _Float16* W2h   = W1h + (size_t)N_EXP * D_EMB * F_HID;        // 36 MB

    moe_zero<<<(T_TOK * D_EMB + 255) / 256, 256, 0, stream>>>(out, T_TOK * D_EMB, cnt);

    // pack weights once per launch (amortized over ~32 tile-reuses per expert)
    {
        const size_t n1 = (size_t)N_EXP * NT1 * KT1 * 32;  // 1,179,648 threads
        const size_t n2 = (size_t)N_EXP * NT2 * KT2 * 32;
        moe_pack_b<<<(unsigned)((n1 + 255) / 256), 256, 0, stream>>>(W1, W1h, D_EMB, F_HID);
        moe_pack_b<<<(unsigned)((n2 + 255) / 256), 256, 0, stream>>>(W2, W2h, F_HID, D_EMB);
    }

    moe_router<<<(T_TOK * 32) / 256, 256, 0, stream>>>(x, noise, Wg, bg, Wn, bn,
                                                       cnt, toks, gates);

    const size_t shmem = (size_t)MT * (XS_STR + HS_STR) * sizeof(_Float16); // ~247 KB
    dim3 grid(T_TOK / MT, N_EXP);
    moe_ffn<<<grid, 256, shmem, stream>>>(x, W1h, b1, W2h, b2,
                                          cnt, toks, gates, out);
}
